// TransformerEncoderLayer_9371618640156
// MI455X (gfx1250) — matrix operand order repaired and checked by reading, not compiled, not run on hardware
//
#include <hip/hip_runtime.h>
#include <hip/hip_bf16.h>

#define BB 2
#define SS 4096
#define DM 512
#define NH 8
#define HD 64
#define DFF 2048
#define MM (BB * SS)   // 8192 rows
#define KC 512         // K-panel staged in LDS per TDM descriptor

typedef __attribute__((ext_vector_type(16))) __bf16 v16bf;
typedef __attribute__((ext_vector_type(8)))  __bf16 v8bf;
typedef __attribute__((ext_vector_type(8)))  float  v8f;
typedef __attribute__((ext_vector_type(4)))  unsigned int u32x4;
typedef __attribute__((ext_vector_type(8)))  unsigned int u32x8;

union V16U { v16bf v; v8bf h[2]; };

__device__ __forceinline__ v16bf make_v16(v8bf lo, v8bf hi) {
  V16U u; u.h[0] = lo; u.h[1] = hi; return u.v;
}

__device__ __forceinline__ __bf16 f2bf(float f) {
  unsigned int u = __builtin_bit_cast(unsigned int, f);
  u += 0x7FFFu + ((u >> 16) & 1u);                  // round-to-nearest-even
  unsigned short h = (unsigned short)(u >> 16);
  return __builtin_bit_cast(__bf16, h);
}

// ---------------------------------------------------------------- converters
__global__ void k_f32_to_bf16(const float* __restrict__ in,
                              __bf16* __restrict__ out, int n) {
  int i = blockIdx.x * blockDim.x + threadIdx.x;
  if (i < n) out[i] = f2bf(in[i]);
}

// Vt[b][h][d][s] = V[b*S+s][h*64+d]
__global__ void k_transpose_v(const __bf16* __restrict__ V,
                              __bf16* __restrict__ Vt, int n) {
  int i = blockIdx.x * blockDim.x + threadIdx.x;
  if (i >= n) return;
  int s = i & (SS - 1);
  int rest = i >> 12;            // /4096
  int d = rest & 63; rest >>= 6;
  int h = rest & 7;  int b = rest >> 3;
  Vt[i] = V[((size_t)(b * SS + s)) * DM + h * HD + d];
}

// ------------------------------------------------- WMMA GEMM: Y = X*W^T + b
// X: [M,K] bf16 row-major, W: [N,K] bf16 row-major (so W^T is [K,N]).
// Block (128 thr = 4 waves) owns a 64(M) x 64(N) tile. The 64x512 weight
// panel for the block's N-strip is DMA'd into LDS by the Tensor Data Mover
// (one tensor_load_to_lds per K-panel, TENSORcnt-tracked), then all 4 waves
// consume it via ds_load fragments while A streams from global.
template <bool RELU>
__global__ void k_gemm_bias(const __bf16* __restrict__ X,
                            const __bf16* __restrict__ W,
                            const float*  __restrict__ bias,
                            float*  __restrict__ outF,   // [M,N] or null
                            __bf16* __restrict__ outB,   // [M,N] or null
                            int M, int N, int K) {
  const int lane = threadIdx.x & 31;
  const int wave = threadIdx.x >> 5;
  const int half = lane >> 4;
  const int r    = lane & 15;
  const int tn0  = blockIdx.x * 64;
  const int tm   = (blockIdx.y * 4 + wave) * 16;

  __shared__ __align__(128) __bf16 Blds[64 * KC];   // 64 KB weight panel

  v8f acc[4];
#pragma unroll
  for (int t = 0; t < 4; ++t) acc[t] = (v8f){0,0,0,0,0,0,0,0};

  const __bf16* Arow = X + (size_t)(tm + r) * K;

  for (int kc = 0; kc < K; kc += KC) {
    __syncthreads();   // previous panel fully consumed before DMA overwrite
    if (threadIdx.x < 32) {
      // --- build Tensor DMA Descriptor (D#): 2-D tensor, 2-byte elements
      unsigned long long ga =
          (unsigned long long)(uintptr_t)(W + (size_t)tn0 * K + kc);
      unsigned int la = (unsigned int)(uintptr_t)(void*)&Blds[0];
      u32x4 g0;
      g0[0] = 1u;                                     // count=1, user mode
      g0[1] = la;                                     // lds_addr
      g0[2] = (unsigned int)ga;                       // global_addr[31:0]
      g0[3] = (unsigned int)(ga >> 32) | 0x80000000u; // addr[56:32] | type=2
      u32x8 g1;
      g1[0] = 1u << 16;                               // data_size = 2 bytes
      g1[1] = ((unsigned)KC & 0xFFFFu) << 16;         // tensor_dim0 lo16
      g1[2] = ((unsigned)KC >> 16) | (64u << 16);     // dim0 hi | tensor_dim1 lo
      g1[3] = ((unsigned)KC) << 16;                   // dim1 hi=0 | tile_dim0
      g1[4] = 64u;                                    // tile_dim1 (tile_dim2=0)
      g1[5] = (unsigned)K;                            // dim0_stride[31:0]
      g1[6] = 0u;                                     // dim0_stride hi | s1 lo
      g1[7] = 0u;
      asm volatile("tensor_load_to_lds %0, %1" :: "s"(g0), "s"(g1) : "memory");
      __builtin_amdgcn_s_wait_tensorcnt(0);
    }
    __syncthreads();   // panel visible to all waves

    for (int kkl = 0; kkl < KC; kkl += 32) {
      const int kk = kc + kkl;
      v8bf lo = *(const v8bf*)(Arow + kk + 16 * half);
      v8bf hi = *(const v8bf*)(Arow + kk + 16 * half + 8);
      v16bf a = make_v16(lo, hi);
#pragma unroll
      for (int t = 0; t < 4; ++t) {
        v16bf b = *(const v16bf*)&Blds[(t * 16 + r) * KC + kkl + 16 * half];
        acc[t] = __builtin_amdgcn_wmma_f32_16x16x32_bf16(
            false, a, false, b, (short)0, acc[t], false, false);
      }
    }
  }

#pragma unroll
  for (int t = 0; t < 4; ++t) {
    int n = tn0 + t * 16 + r;
    float bv = bias[n];
#pragma unroll
    for (int v = 0; v < 8; ++v) {
      int m = tm + v + 8 * half;
      float val = acc[t][v] + bv;
      if (RELU) val = fmaxf(val, 0.0f);
      size_t idx = (size_t)m * N + n;
      if (outF) outF[idx] = val;
      if (outB) outB[idx] = f2bf(val);
    }
  }
}

// ----------------------------------------------------- flash attention wave
// One wave per (b, h, 16-query tile). Q[16x64] @ K^T, online softmax, @ V.
__global__ void k_flash_attn(const __bf16* __restrict__ Q,   // [M, 512]
                             const __bf16* __restrict__ Km,  // [M, 512]
                             const __bf16* __restrict__ Vt,  // [B][H][64][S]
                             const float*  __restrict__ mask,// [B,S,S]
                             __bf16* __restrict__ ctx) {     // [M, 512]
  const int lane = threadIdx.x & 31;
  const int half = lane >> 4;
  const int r    = lane & 15;
  const int q0   = blockIdx.x * 16;
  const int h    = blockIdx.y;
  const int b    = blockIdx.z;

  __shared__ __align__(32) __bf16 Pl[16 * 32];

  // Q fragments for the two 32-wide K steps of the 64-dim head
  const __bf16* Qrow = Q + (size_t)(b * SS + q0 + r) * DM + h * HD;
  v16bf qa[2];
#pragma unroll
  for (int kk = 0; kk < 2; ++kk) {
    v8bf lo = *(const v8bf*)(Qrow + kk * 32 + 16 * half);
    v8bf hi = *(const v8bf*)(Qrow + kk * 32 + 16 * half + 8);
    qa[kk] = make_v16(lo, hi);
  }

  v8f ctxacc[4];
#pragma unroll
  for (int t = 0; t < 4; ++t) ctxacc[t] = (v8f){0,0,0,0,0,0,0,0};
  float Mrun[8], Lrun[8];
#pragma unroll
  for (int v = 0; v < 8; ++v) { Mrun[v] = -3.0e38f; Lrun[v] = 0.0f; }

  const float scale = 0.125f;  // 1/sqrt(64)
  const __bf16* Vbase = Vt + ((size_t)(b * NH + h)) * HD * SS;

  for (int j0 = 0; j0 < SS; j0 += 32) {
    // ---- scores: two 16-key tiles, K-dim = 64
    v8f sc[2];
#pragma unroll
    for (int t = 0; t < 2; ++t) {
      sc[t] = (v8f){0,0,0,0,0,0,0,0};
      const __bf16* Krow = Km + (size_t)(b * SS + j0 + t * 16 + r) * DM + h * HD;
#pragma unroll
      for (int kk = 0; kk < 2; ++kk) {
        v16bf kb = *(const v16bf*)(Krow + kk * 32 + 16 * half);
        sc[t] = __builtin_amdgcn_wmma_f32_16x16x32_bf16(
            false, qa[kk], false, kb, (short)0, sc[t], false, false);
      }
    }

    // ---- online softmax over the 32 new columns
    float p0a[8], p1a[8];
#pragma unroll
    for (int v = 0; v < 8; ++v) {
      int m = v + 8 * half;  // query row within tile
      const float* mr = mask + ((size_t)b * SS + q0 + m) * SS + j0;
      float s0 = sc[0][v] * scale + (1.0f - mr[r])      * -1e9f;
      float s1 = sc[1][v] * scale + (1.0f - mr[16 + r]) * -1e9f;
      float mx = fmaxf(s0, s1);
#pragma unroll
      for (int off = 1; off < 16; off <<= 1)
        mx = fmaxf(mx, __shfl_xor(mx, off, 32));
      float Mnew  = fmaxf(Mrun[v], mx);
      float alpha = __expf(Mrun[v] - Mnew);
      float p0 = __expf(s0 - Mnew);
      float p1 = __expf(s1 - Mnew);
      float sum = p0 + p1;
#pragma unroll
      for (int off = 1; off < 16; off <<= 1)
        sum += __shfl_xor(sum, off, 32);
      Lrun[v] = Lrun[v] * alpha + sum;
      Mrun[v] = Mnew;
      p0a[v] = p0; p1a[v] = p1;
#pragma unroll
      for (int t = 0; t < 4; ++t) ctxacc[t][v] *= alpha;
    }

    // ---- re-layout P (C-layout -> A-layout) via LDS
    __syncthreads();
#pragma unroll
    for (int v = 0; v < 8; ++v) {
      int m = v + 8 * half;
      Pl[m * 32 + r]      = f2bf(p0a[v]);
      Pl[m * 32 + 16 + r] = f2bf(p1a[v]);
    }
    __syncthreads();
    v8bf plo = *(const v8bf*)&Pl[r * 32 + 16 * half];
    v8bf phi = *(const v8bf*)&Pl[r * 32 + 16 * half + 8];
    v16bf pa = make_v16(plo, phi);

    // ---- ctx += P[16x32] @ V[32x64]  (B-frags contiguous from V^T)
#pragma unroll
    for (int t = 0; t < 4; ++t) {
      int d = t * 16 + r;
      v16bf vb = *(const v16bf*)(Vbase + (size_t)d * SS + j0 + 16 * half);
      ctxacc[t] = __builtin_amdgcn_wmma_f32_16x16x32_bf16(
          false, pa, false, vb, (short)0, ctxacc[t], false, false);
    }
  }

  // ---- epilogue: normalize and store bf16 context
#pragma unroll
  for (int t = 0; t < 4; ++t) {
#pragma unroll
    for (int v = 0; v < 8; ++v) {
      int m = v + 8 * half;
      float val = ctxacc[t][v] / Lrun[v];
      ctx[(size_t)(b * SS + q0 + m) * DM + h * HD + t * 16 + r] = f2bf(val);
    }
  }
}

// -------------------------------------------- residual add + LayerNorm (512)
__global__ void k_add_ln(const float* __restrict__ y,
                         const float* __restrict__ resid,
                         const float* __restrict__ g,
                         const float* __restrict__ bt,
                         float*  __restrict__ outF,
                         __bf16* __restrict__ outB) {
  const int row = blockIdx.x, tid = threadIdx.x;  // 128 threads
  __shared__ float red[128];
  const float* yr = y + (size_t)row * DM;
  const float* rr = resid + (size_t)row * DM;
  float v[4]; float s = 0.0f;
#pragma unroll
  for (int i = 0; i < 4; ++i) {
    int c = tid + i * 128;
    v[i] = yr[c] + rr[c];
    s += v[i];
  }
  red[tid] = s; __syncthreads();
  for (int o = 64; o > 0; o >>= 1) {
    if (tid < o) red[tid] += red[tid + o];
    __syncthreads();
  }
  float mu = red[0] * (1.0f / 512.0f);
  __syncthreads();
  float sv = 0.0f;
#pragma unroll
  for (int i = 0; i < 4; ++i) { float d = v[i] - mu; sv += d * d; }
  red[tid] = sv; __syncthreads();
  for (int o = 64; o > 0; o >>= 1) {
    if (tid < o) red[tid] += red[tid + o];
    __syncthreads();
  }
  float inv = rsqrtf(red[0] * (1.0f / 512.0f) + 1e-5f);
#pragma unroll
  for (int i = 0; i < 4; ++i) {
    int c = tid + i * 128;
    float o = (v[i] - mu) * inv * g[c] + bt[c];
    size_t idx = (size_t)row * DM + c;
    outF[idx] = o;
    if (outB) outB[idx] = f2bf(o);
  }
}

// ---------------------------------------------------------------------------
extern "C" void kernel_launch(void* const* d_in, const int* in_sizes, int n_in,
                              void* d_out, int out_size, void* d_ws, size_t ws_size,
                              hipStream_t stream) {
  (void)in_sizes; (void)n_in; (void)out_size; (void)ws_size;
  const float* src  = (const float*)d_in[0];
  const float* mask = (const float*)d_in[1];
  const float* Wq = (const float*)d_in[2];  const float* bq = (const float*)d_in[3];
  const float* Wk = (const float*)d_in[4];  const float* bk = (const float*)d_in[5];
  const float* Wv = (const float*)d_in[6];  const float* bv = (const float*)d_in[7];
  const float* Wo = (const float*)d_in[8];  const float* bo = (const float*)d_in[9];
  const float* W1 = (const float*)d_in[10]; const float* b1 = (const float*)d_in[11];
  const float* W2 = (const float*)d_in[12]; const float* b2 = (const float*)d_in[13];
  const float* ln1g = (const float*)d_in[14]; const float* ln1b = (const float*)d_in[15];
  const float* ln2g = (const float*)d_in[16]; const float* ln2b = (const float*)d_in[17];
  float* out = (float*)d_out;

  char* w = (char*)d_ws; size_t off = 0;
  auto alloc = [&](size_t bytes) -> void* {
    void* p = w + off; off = (off + bytes + 255) & ~(size_t)255; return p;
  };
  __bf16* Xb   = (__bf16*)alloc((size_t)MM * DM * 2);
  __bf16* Wqb  = (__bf16*)alloc((size_t)DM * DM * 2);
  __bf16* Wkb  = (__bf16*)alloc((size_t)DM * DM * 2);
  __bf16* Wvb  = (__bf16*)alloc((size_t)DM * DM * 2);
  __bf16* Wob  = (__bf16*)alloc((size_t)DM * DM * 2);
  __bf16* W1b  = (__bf16*)alloc((size_t)DFF * DM * 2);
  __bf16* W2b  = (__bf16*)alloc((size_t)DM * DFF * 2);
  __bf16* Qb   = (__bf16*)alloc((size_t)MM * DM * 2);
  __bf16* Kb   = (__bf16*)alloc((size_t)MM * DM * 2);
  __bf16* Vb   = (__bf16*)alloc((size_t)MM * DM * 2);
  __bf16* Vtb  = (__bf16*)alloc((size_t)MM * DM * 2);
  __bf16* Ctxb = (__bf16*)alloc((size_t)MM * DM * 2);
  float*  SAf  = (float*) alloc((size_t)MM * DM * 4);   // reused for FFN out
  float*  X1f  = (float*) alloc((size_t)MM * DM * 4);
  __bf16* X1b  = (__bf16*)alloc((size_t)MM * DM * 2);
  __bf16* FFHb = (__bf16*)alloc((size_t)MM * DFF * 2);

  auto cvt = [&](const float* in, __bf16* o, int n) {
    k_f32_to_bf16<<<(n + 255) / 256, 256, 0, stream>>>(in, o, n);
  };
  cvt(src, Xb, MM * DM);
  cvt(Wq, Wqb, DM * DM); cvt(Wk, Wkb, DM * DM);
  cvt(Wv, Wvb, DM * DM); cvt(Wo, Wob, DM * DM);
  cvt(W1, W1b, DFF * DM); cvt(W2, W2b, DM * DFF);

  const dim3 gemm_blk(128);
  // Q/K/V projections  ([8192,512] x [512,512]^T), bf16 outputs
  k_gemm_bias<false><<<dim3(DM / 64, MM / 64), gemm_blk, 0, stream>>>(
      Xb, Wqb, bq, nullptr, Qb, MM, DM, DM);
  k_gemm_bias<false><<<dim3(DM / 64, MM / 64), gemm_blk, 0, stream>>>(
      Xb, Wkb, bk, nullptr, Kb, MM, DM, DM);
  k_gemm_bias<false><<<dim3(DM / 64, MM / 64), gemm_blk, 0, stream>>>(
      Xb, Wvb, bv, nullptr, Vb, MM, DM, DM);

  // V^T for contiguous B-fragments in attention
  {
    int n = BB * NH * HD * SS;
    k_transpose_v<<<(n + 255) / 256, 256, 0, stream>>>(Vb, Vtb, n);
  }

  // Streaming (flash) attention: 1 wave per (b, h, 16-query tile)
  k_flash_attn<<<dim3(SS / 16, NH, BB), dim3(32), 0, stream>>>(
      Qb, Kb, Vtb, mask, Ctxb);

  // Output projection -> f32
  k_gemm_bias<false><<<dim3(DM / 64, MM / 64), gemm_blk, 0, stream>>>(
      Ctxb, Wob, bo, SAf, nullptr, MM, DM, DM);

  // residual + LN1 -> X1 (f32 + bf16)
  k_add_ln<<<MM, 128, 0, stream>>>(SAf, src, ln1g, ln1b, X1f, X1b);

  // FFN: relu(X1 @ W1^T + b1) -> bf16 hidden
  k_gemm_bias<true><<<dim3(DFF / 64, MM / 64), gemm_blk, 0, stream>>>(
      X1b, W1b, b1, nullptr, FFHb, MM, DFF, DM);
  // hidden @ W2^T + b2 -> f32 (reuse SAf)
  k_gemm_bias<false><<<dim3(DM / 64, MM / 64), gemm_blk, 0, stream>>>(
      FFHb, W2b, b2, SAf, nullptr, MM, DM, DFF);

  // residual + LN2 -> d_out
  k_add_ln<<<MM, 128, 0, stream>>>(SAf, X1f, ln2g, ln2b, out, nullptr);
}
